// TSP_EdgeGNN_15908558865690
// MI455X (gfx1250) — compile-verified, matrix operand order checked by reading, not run on hardware
//
#include <hip/hip_runtime.h>
#include <hip/hip_bf16.h>
#include <hip/hip_fp16.h>

#define HEADS 4
#define HID 64
#define NEG_SLOPE 0.2f

typedef __attribute__((ext_vector_type(16))) _Float16 v16h;
typedef __attribute__((ext_vector_type(8)))  _Float16 v8h;
typedef __attribute__((ext_vector_type(8)))  float    v8f;

// ---------------------------------------------------------------------------
// WMMA helpers (gfx1250: V_WMMA_F32_16X16X32_F16)
// A fragment (16x32 f16): lane = (half, lr); lane holds K = half*8 + {0..7}
// in VGPRs 0..3 and K = 16 + half*8 + {0..7} in VGPRs 4..7, row M = lr.
// B fragment (32x16 f16): mirrored, column N = lr, same K packing -> we store
// weights pre-transposed [N][K] so both loads are two contiguous 16B reads.
// ---------------------------------------------------------------------------
__device__ __forceinline__ v16h frag_load64(const _Float16* rowbase, int k0, int half) {
  v8h lo = *(const v8h*)(rowbase + k0 + half * 8);
  v8h hi = *(const v8h*)(rowbase + k0 + 16 + half * 8);
  return __builtin_shufflevector(lo, hi, 0,1,2,3,4,5,6,7,8,9,10,11,12,13,14,15);
}

__device__ __forceinline__ v8f wmma_f16(v16h a, v16h b, v8f c) {
  return __builtin_amdgcn_wmma_f32_16x16x32_f16(false, a, false, b, (short)0, c,
                                                false, false);
}

// Monotone float<->uint encoding for atomicMax-based segment max.
__device__ __forceinline__ unsigned fenc(float f) {
  unsigned u = __float_as_uint(f);
  return (u & 0x80000000u) ? ~u : (u | 0x80000000u);
}
__device__ __forceinline__ float fdec(unsigned u) {
  return __uint_as_float((u & 0x80000000u) ? (u & 0x7FFFFFFFu) : ~u);
}

// ---------------------------------------------------------------------------
// One-time weight conversion: f32 -> transposed f16
// ---------------------------------------------------------------------------
__global__ void k_cvt(const float* __restrict__ Wl, const float* __restrict__ Wr,
                      const float* __restrict__ W1, const float* __restrict__ W2,
                      _Float16* __restrict__ WlT, _Float16* __restrict__ WrT,
                      _Float16* __restrict__ W1sT, _Float16* __restrict__ W1dT,
                      _Float16* __restrict__ W2T) {
  int id = blockIdx.x * 256 + threadIdx.x;
  if (id < 3 * 256 * 64) {                       // Wl/Wr: [3,64,256] -> [3,256,64]
    int l = id / (256 * 64), rem = id % (256 * 64);
    int n = rem / 64, k = rem % 64;
    WlT[id] = (_Float16)Wl[(l * 64 + k) * 256 + n];
    WrT[id] = (_Float16)Wr[(l * 64 + k) * 256 + n];
  }
  int j = id - 3 * 256 * 64;
  if (j >= 0 && j < 64 * 64) {                   // W1 rows 0..63 / 64..127
    int n = j / 64, k = j % 64;
    W1sT[j] = (_Float16)W1[k * 64 + n];
    W1dT[j] = (_Float16)W1[(64 + k) * 64 + n];
  }
  int j2 = j - 64 * 64;
  if (j2 >= 0 && j2 < 32 * 64) {                 // W2: [64,32] -> [32,64]
    int n = j2 / 64, k = j2 % 64;
    W2T[j2] = (_Float16)W2[k * 32 + n];
  }
}

// h = relu(x @ W_emb + b_emb), stored f16 (x is [N,2] -> scalar math)
__global__ void k_embed(const float* __restrict__ x, const float* __restrict__ W_emb,
                        const float* __restrict__ b_emb, _Float16* __restrict__ h,
                        int N) {
  int id = blockIdx.x * 256 + threadIdx.x;
  if (id >= N * 64) return;
  int n = id >> 6, c = id & 63;
  float v = x[n * 2] * W_emb[c] + x[n * 2 + 1] * W_emb[64 + c] + b_emb[c];
  h[id] = (_Float16)(v > 0.f ? v : 0.f);
}

// WMMA GEMM: out[N,256] = h[N,64] @ W[64,256] + bias. One wave per 16x64 tile.
__global__ void k_xform(const _Float16* __restrict__ h, const _Float16* __restrict__ WT,
                        const float* __restrict__ bias, _Float16* __restrict__ out,
                        int N) {
  int lane = threadIdx.x & 31;
  int half = lane >> 4, lr = lane & 15;
  int m0 = blockIdx.x * 16;
  int nbase = blockIdx.y * 64;
  int row = m0 + lr; if (row >= N) row = N - 1;
  const _Float16* arow = h + (size_t)row * 64;
  v16h a0 = frag_load64(arow, 0, half);
  v16h a1 = frag_load64(arow, 32, half);
  const bool full = (m0 + 16 <= N);
#pragma unroll
  for (int nt = 0; nt < 4; ++nt) {
    int n0 = nbase + nt * 16;
    const _Float16* brow = WT + (size_t)(n0 + lr) * 64;
    v8f acc = {};
    acc = wmma_f16(a0, frag_load64(brow, 0, half), acc);
    acc = wmma_f16(a1, frag_load64(brow, 32, half), acc);
    float bv = bias[n0 + lr];
    _Float16* op = out + (size_t)(m0 + half * 8) * 256 + n0 + lr;
    if (full) {
#pragma unroll
      for (int r = 0; r < 8; ++r) op[(size_t)r * 256] = (_Float16)(acc[r] + bv);
    } else {
#pragma unroll
      for (int r = 0; r < 8; ++r) {
        int m = m0 + half * 8 + r;
        if (m < N) op[(size_t)r * 256] = (_Float16)(acc[r] + bv);
      }
    }
  }
}

__global__ void k_init(float* __restrict__ hnew, unsigned* __restrict__ mx,
                       float* __restrict__ den, int N) {
  int id = blockIdx.x * 256 + threadIdx.x;
  if (id >= N * 64) return;
  hnew[id] = 0.f;
  int c = id & 63;
  if (c < 4) { int n = id >> 6; mx[n * 4 + c] = 0x007FFFFFu; den[n * 4 + c] = 0.f; }
}

// Pass 1: per (edge, head) attention logit + segment max via encoded atomicMax
__global__ void k_logits(const _Float16* __restrict__ xl, const _Float16* __restrict__ xr,
                         const int* __restrict__ src, const int* __restrict__ dst,
                         const float* __restrict__ eattr, const float* __restrict__ We_l,
                         const float* __restrict__ att_l, float* __restrict__ logits,
                         unsigned* __restrict__ mx, int E) {
  __shared__ float sWe[256], sAtt[256];
  int tid = threadIdx.x;
  sWe[tid] = We_l[tid];
  sAtt[tid] = att_l[tid];
  __syncthreads();
  int gid = blockIdx.x * 256 + tid;
  if (gid >= E * 4) return;
  int e = gid >> 2, hh = gid & 3;
  int s = src[e], d = dst[e];
  float ea = eattr[e];
  const _Float16* pl = xl + (size_t)s * 256 + hh * 64;
  const _Float16* pr = xr + (size_t)d * 256 + hh * 64;
  float logit = 0.f;
#pragma unroll
  for (int cb = 0; cb < 64; cb += 8) {
    v8h a = *(const v8h*)(pl + cb);
    v8h b = *(const v8h*)(pr + cb);
#pragma unroll
    for (int j = 0; j < 8; ++j) {
      float m = (float)a[j] + (float)b[j] + ea * sWe[hh * 64 + cb + j];
      m = m > 0.f ? m : NEG_SLOPE * m;
      logit += m * sAtt[hh * 64 + cb + j];
    }
  }
  logits[gid] = logit;
  atomicMax(mx + d * 4 + hh, fenc(logit));
}

// Pass 2: a = exp(logit - max); accumulate denominator
__global__ void k_exp(float* __restrict__ logits, const int* __restrict__ dst,
                      const unsigned* __restrict__ mx, float* __restrict__ den, int E) {
  int gid = blockIdx.x * 256 + threadIdx.x;
  if (gid >= E * 4) return;
  int e = gid >> 2, hh = gid & 3;
  int d = dst[e];
  float a = __expf(logits[gid] - fdec(mx[d * 4 + hh]));
  logits[gid] = a;
  atomicAdd(den + d * 4 + hh, a);
}

// Pass 3: hnew[dst] += mean_h(alpha_h * xl[src][h]) (head mean folded in)
__global__ void k_aggr(const _Float16* __restrict__ xl, const int* __restrict__ src,
                       const int* __restrict__ dst, const float* __restrict__ alph,
                       const float* __restrict__ den, float* __restrict__ hnew, int E) {
  int gid = blockIdx.x * 256 + threadIdx.x;
  if (gid >= E * 64) return;
  int e = gid >> 6, c = gid & 63;
  int s = src[e], d = dst[e];
  const _Float16* p = xl + (size_t)s * 256 + c;
  float acc = 0.f;
#pragma unroll
  for (int hh = 0; hh < 4; ++hh) {
    float alpha = alph[e * 4 + hh] / (den[d * 4 + hh] + 1e-16f);
    acc += alpha * (float)p[hh * 64];
  }
  atomicAdd(hnew + (size_t)d * 64 + c, 0.25f * acc);
}

__global__ void k_finish(const float* __restrict__ hnew, const float* __restrict__ cbias,
                         _Float16* __restrict__ h, int N, int doRelu) {
  int id = blockIdx.x * 256 + threadIdx.x;
  if (id >= N * 64) return;
  float v = hnew[id] + cbias[id & 63];
  if (doRelu) v = v > 0.f ? v : 0.f;
  h[id] = (_Float16)v;
}

// Fused edge MLP: one wave per 16-edge tile; gathered-row WMMA for layer 1,
// LDS re-fragmentation, WMMA layer 2, cross-lane reduction for [32,1] head.
__global__ void k_mlp(const _Float16* __restrict__ h, const int* __restrict__ src,
                      const int* __restrict__ dst, const float* __restrict__ eattr,
                      const _Float16* __restrict__ W1sT, const _Float16* __restrict__ W1dT,
                      const float* __restrict__ W1, const float* __restrict__ b1,
                      const _Float16* __restrict__ W2T, const float* __restrict__ b2,
                      const float* __restrict__ W3, const float* __restrict__ b3,
                      float* __restrict__ out, int E) {
  __shared__ __align__(16) _Float16 zlds[16 * 72];   // stride 72 (144B, 16B aligned)
  __shared__ float plds[16][16];
  int lane = threadIdx.x & 31;
  int half = lane >> 4, lr = lane & 15;
  int tile = blockIdx.x;
  const bool full = (tile * 16 + 16 <= E);
  int ea_idx = tile * 16 + lr;
  int ec = ea_idx < E ? ea_idx : E - 1;
  const _Float16* sp = h + (size_t)src[ec] * 64;
  const _Float16* dp = h + (size_t)dst[ec] * 64;
  v16h as0 = frag_load64(sp, 0, half);
  v16h as1 = frag_load64(sp, 32, half);
  v16h ad0 = frag_load64(dp, 0, half);
  v16h ad1 = frag_load64(dp, 32, half);
  const float* w1e = W1 + 128 * 64;
  const float* eap = eattr + tile * 16 + half * 8;
#pragma unroll
  for (int nt = 0; nt < 4; ++nt) {
    int n0 = nt * 16;
    const _Float16* bs = W1sT + (size_t)(n0 + lr) * 64;
    const _Float16* bd = W1dT + (size_t)(n0 + lr) * 64;
    v8f acc = {};
    acc = wmma_f16(as0, frag_load64(bs, 0, half), acc);
    acc = wmma_f16(as1, frag_load64(bs, 32, half), acc);
    acc = wmma_f16(ad0, frag_load64(bd, 0, half), acc);
    acc = wmma_f16(ad1, frag_load64(bd, 32, half), acc);
    float bb = b1[n0 + lr];
    float we = w1e[n0 + lr];
    if (full) {
#pragma unroll
      for (int r = 0; r < 8; ++r) {
        float v = acc[r] + bb + eap[r] * we;
        zlds[(half * 8 + r) * 72 + n0 + lr] = (_Float16)(v > 0.f ? v : 0.f);
      }
    } else {
#pragma unroll
      for (int r = 0; r < 8; ++r) {
        int e = tile * 16 + half * 8 + r;
        float ea = eattr[e < E ? e : E - 1];
        float v = acc[r] + bb + ea * we;
        zlds[(half * 8 + r) * 72 + n0 + lr] = (_Float16)(v > 0.f ? v : 0.f);
      }
    }
  }
  __syncthreads();
  const _Float16* zrow = zlds + lr * 72;
  v16h az0 = frag_load64(zrow, 0, half);
  v16h az1 = frag_load64(zrow, 32, half);
  v8f acc2[2];
#pragma unroll
  for (int nt = 0; nt < 2; ++nt) {
    const _Float16* bw = W2T + (size_t)(nt * 16 + lr) * 64;
    v8f acc = {};
    acc = wmma_f16(az0, frag_load64(bw, 0, half), acc);
    acc = wmma_f16(az1, frag_load64(bw, 32, half), acc);
    acc2[nt] = acc;
  }
  float w3a = W3[lr], w3b = W3[16 + lr];
  float b2a = b2[lr], b2b = b2[16 + lr];
#pragma unroll
  for (int r = 0; r < 8; ++r) {
    float v0 = acc2[0][r] + b2a; v0 = v0 > 0.f ? v0 : 0.f;
    float v1 = acc2[1][r] + b2b; v1 = v1 > 0.f ? v1 : 0.f;
    plds[half * 8 + r][lr] = v0 * w3a + v1 * w3b;
  }
  __syncthreads();
  if (lane < 16) {
    float s = b3[0];
#pragma unroll
    for (int c = 0; c < 16; ++c) s += plds[lane][c];
    int e = tile * 16 + lane;
    if (full) out[e] = s;
    else if (e < E) out[e] = s;
  }
}

extern "C" void kernel_launch(void* const* d_in, const int* in_sizes, int n_in,
                              void* d_out, int out_size, void* d_ws, size_t ws_size,
                              hipStream_t stream) {
  const float* x     = (const float*)d_in[0];
  const int*   eidx  = (const int*)d_in[1];
  const float* eattr = (const float*)d_in[2];
  const float* W_emb = (const float*)d_in[3];
  const float* b_emb = (const float*)d_in[4];
  const float* Wl    = (const float*)d_in[5];
  const float* bl    = (const float*)d_in[6];
  const float* Wr    = (const float*)d_in[7];
  const float* br    = (const float*)d_in[8];
  const float* We    = (const float*)d_in[9];
  const float* att   = (const float*)d_in[10];
  const float* cbias = (const float*)d_in[11];
  const float* W1    = (const float*)d_in[12];
  const float* b1    = (const float*)d_in[13];
  const float* W2    = (const float*)d_in[14];
  const float* b2    = (const float*)d_in[15];
  const float* W3    = (const float*)d_in[16];
  const float* b3    = (const float*)d_in[17];

  const int N = in_sizes[0] / 2;
  const int E = in_sizes[2];
  const int* src = eidx;
  const int* dst = eidx + E;

  char* ws = (char*)d_ws;
  size_t off = 0;
  auto alloc = [&](size_t bytes) {
    off = (off + 255) & ~(size_t)255;
    char* p = ws + off;
    off += bytes;
    return p;
  };
  _Float16* h      = (_Float16*)alloc((size_t)N * 64 * 2);
  _Float16* xl     = (_Float16*)alloc((size_t)N * 256 * 2);
  _Float16* xr     = (_Float16*)alloc((size_t)N * 256 * 2);
  float*    hnew   = (float*)   alloc((size_t)N * 64 * 4);
  unsigned* mx     = (unsigned*)alloc((size_t)N * 4 * 4);
  float*    den    = (float*)   alloc((size_t)N * 4 * 4);
  float*    logits = (float*)   alloc((size_t)E * 4 * 4);
  _Float16* WlT    = (_Float16*)alloc(3 * 256 * 64 * 2);
  _Float16* WrT    = (_Float16*)alloc(3 * 256 * 64 * 2);
  _Float16* W1sT   = (_Float16*)alloc(64 * 64 * 2);
  _Float16* W1dT   = (_Float16*)alloc(64 * 64 * 2);
  _Float16* W2T    = (_Float16*)alloc(32 * 64 * 2);

  const int cvt_jobs = 3 * 256 * 64 + 64 * 64 + 32 * 64;
  k_cvt<<<(cvt_jobs + 255) / 256, 256, 0, stream>>>(Wl, Wr, W1, W2, WlT, WrT,
                                                    W1sT, W1dT, W2T);
  k_embed<<<(N * 64 + 255) / 256, 256, 0, stream>>>(x, W_emb, b_emb, h, N);

  const int mtiles = (N + 15) / 16;
  for (int layer = 0; layer < 3; ++layer) {
    dim3 gx(mtiles, 4);
    k_xform<<<gx, 32, 0, stream>>>(h, WlT + layer * 16384, bl + layer * 256, xl, N);
    k_xform<<<gx, 32, 0, stream>>>(h, WrT + layer * 16384, br + layer * 256, xr, N);
    k_init<<<(N * 64 + 255) / 256, 256, 0, stream>>>(hnew, mx, den, N);
    k_logits<<<(E * 4 + 255) / 256, 256, 0, stream>>>(xl, xr, src, dst, eattr,
                                                      We + layer * 256,
                                                      att + layer * 256, logits, mx, E);
    k_exp<<<(E * 4 + 255) / 256, 256, 0, stream>>>(logits, dst, mx, den, E);
    k_aggr<<<(E * 64 + 255) / 256, 256, 0, stream>>>(xl, src, dst, logits, den, hnew, E);
    k_finish<<<(N * 64 + 255) / 256, 256, 0, stream>>>(hnew, cbias + layer * 64, h, N,
                                                       layer < 2 ? 1 : 0);
  }
  k_mlp<<<(E + 15) / 16, 32, 0, stream>>>(h, src, dst, eattr, W1sT, W1dT, W1, b1,
                                          W2T, b2, W3, b3, (float*)d_out, E);
}
